// PPO_36988258353766
// MI455X (gfx1250) — matrix operand order repaired and checked by reading, not compiled, not run on hardware
//
#include <hip/hip_runtime.h>

// ---------------- PPO constants (match reference) ----------------
#define GAMMA_F 0.99f
#define LAM_F   0.95f
#define CLIP_F  0.2f
#define SDIM 128   // state dim
#define HDIM 64    // hidden dim
#define ADIM 16    // action dim
#define LDX  136   // padded LDS row stride (halfwords) for 128-wide X tiles
#define LDH  72    // padded LDS row stride (halfwords) for 64-wide H tiles
#define WAVES 4    // waves per block (wave32)
#define SEG  256   // GAE scan segment length
#define GA   1024  // actor grid
#define GC   1024  // critic grid
#define RB   512   // adv reduction blocks
#define RB2  512   // actor-loss reduction blocks

typedef __attribute__((ext_vector_type(16))) __bf16 v16bf;
typedef __attribute__((ext_vector_type(8)))  float  v8f;

// f32 -> bf16 bits, round-to-nearest-even
__device__ __forceinline__ unsigned short f2bf(float f) {
  unsigned int u = __float_as_uint(f);
  u += 0x7FFFu + ((u >> 16) & 1u);
  return (unsigned short)(u >> 16);
}

__device__ __forceinline__ v8f wmma_bf16(v16bf a, v16bf b, v8f c) {
  // D = A(16x32 bf16) x B(32x16 bf16) + C(16x16 f32)
  return __builtin_amdgcn_wmma_f32_16x16x32_bf16(false, a, false, b, (short)0, c,
                                                 false, false);
}

// A-operand loader from row-major bf16 LDS tile (16 rows x ld halfwords).
// ISA layout (16-bit A 16x32): lanes 0-15 hold rows 0-15 with K = k0+[0..7] and
// k0+[16..23]; lanes 16-31 hold the same rows with K = k0+[8..15], k0+[24..31].
__device__ __forceinline__ v16bf load_a(const unsigned short* buf, int ld, int k0) {
  int lane = threadIdx.x & 31;
  int row = lane & 15;
  int ks = k0 + ((lane >> 4) << 3);
  union { uint4 q[2]; v16bf v; } u;
  u.q[0] = *(const uint4*)(buf + row * ld + ks);
  u.q[1] = *(const uint4*)(buf + row * ld + ks + 16);
  return u.v;
}

// B-operand loader: fragment pre-swizzled in LDS as [lane][16] halfwords.
__device__ __forceinline__ v16bf load_b(const unsigned short* frag) {
  int lane = threadIdx.x & 31;
  union { uint4 q[2]; v16bf v; } u;
  const uint4* p = (const uint4*)(frag + lane * 16);
  u.q[0] = p[0];
  u.q[1] = p[1];
  return u.v;
}

// Swizzle a KxN (row-major f32) weight matrix into WMMA B fragments:
// frag(kc,nt): lane ln, element e  ->  K = kc*32 + (ln>=16)*8 + (e&7) + (e>=8)*16, N = nt*16 + (ln&15)
__device__ __forceinline__ void prep_frags(const float* W, unsigned short* frag,
                                           int kchunks, int ntiles, int ldw,
                                           int tid, int nthreads) {
  int total = kchunks * ntiles * 512;
  for (int i = tid; i < total; i += nthreads) {
    int e = i & 15;
    int ln = (i >> 4) & 31;
    int f = i >> 9;
    int nt = f % ntiles;
    int kc = f / ntiles;
    int K = kc * 32 + ((ln >> 4) << 3) + (e & 7) + ((e >> 3) << 4);
    int N = nt * 16 + (ln & 15);
    frag[i] = f2bf(W[K * ldw + N]);
  }
}

// Stage a 16x128 f32 tile from global into bf16 LDS (coalesced float4 loads).
__device__ __forceinline__ void stage_rows(const float* X, int t0, int T,
                                           unsigned short* dst) {
  int lane = threadIdx.x & 31;
  #pragma unroll
  for (int r = 0; r < 16; ++r) {
    int t = t0 + r;
    float4 x = make_float4(0.f, 0.f, 0.f, 0.f);
    if (t < T) x = *(const float4*)(X + (size_t)t * SDIM + lane * 4);
    uint2 pk;
    pk.x = (unsigned int)f2bf(x.x) | ((unsigned int)f2bf(x.y) << 16);
    pk.y = (unsigned int)f2bf(x.z) | ((unsigned int)f2bf(x.w) << 16);
    *(uint2*)(dst + r * LDX + lane * 4) = pk;
  }
}

__device__ __forceinline__ double block_reduce_d(double v, double* sbuf) {
  int tid = threadIdx.x;
  __syncthreads();
  sbuf[tid] = v;
  __syncthreads();
  for (int s = blockDim.x >> 1; s > 0; s >>= 1) {
    if (tid < (int)s) sbuf[tid] += sbuf[tid + s];
    __syncthreads();
  }
  return sbuf[0];
}

// ---------------- critic: v(s) and v(s') via WMMA MLP ----------------
__device__ __forceinline__ void critic_tile(
    const float* X, float* out, int t0, int T,
    const unsigned short* fW1, const unsigned short* fW2,
    const float* sb1, const float* sb2, const float* sw3, float b3,
    unsigned short* sXw, unsigned short* sH1w) {
  int n = (threadIdx.x & 31) & 15;
  int hi = (threadIdx.x & 31) >> 4;
  stage_rows(X, t0, T, sXw);
  v16bf aX[4];
  #pragma unroll
  for (int kc = 0; kc < 4; ++kc) aX[kc] = load_a(sXw, LDX, kc * 32);
  // L1: H1 = relu(X @ W1 + b1)
  #pragma unroll
  for (int nt = 0; nt < 4; ++nt) {
    v8f acc = {};
    #pragma unroll
    for (int kc = 0; kc < 4; ++kc)
      acc = wmma_bf16(aX[kc], load_b(fW1 + (kc * 4 + nt) * 512), acc);
    float b = sb1[nt * 16 + n];
    #pragma unroll
    for (int r = 0; r < 8; ++r)
      sH1w[(r + hi * 8) * LDH + nt * 16 + n] = f2bf(fmaxf(acc[r] + b, 0.f));
  }
  v16bf aH[2];
  #pragma unroll
  for (int kc = 0; kc < 2; ++kc) aH[kc] = load_a(sH1w, LDH, kc * 32);
  // L2 + fused L3 dot with w3 (64 -> 1) folded into the C-layout registers
  float partial[8] = {};
  #pragma unroll
  for (int nt = 0; nt < 4; ++nt) {
    v8f acc = {};
    #pragma unroll
    for (int kc = 0; kc < 2; ++kc)
      acc = wmma_bf16(aH[kc], load_b(fW2 + (kc * 4 + nt) * 512), acc);
    float b = sb2[nt * 16 + n];
    float w = sw3[nt * 16 + n];
    #pragma unroll
    for (int r = 0; r < 8; ++r)
      partial[r] += fmaxf(acc[r] + b, 0.f) * w;
  }
  #pragma unroll
  for (int r = 0; r < 8; ++r) {
    float s = partial[r];
    #pragma unroll
    for (int m = 1; m <= 8; m <<= 1) s += __shfl_xor(s, m, 32);
    int t = t0 + r + hi * 8;
    if (n == 0 && t < T) out[t] = s + b3;
  }
}

__global__ __launch_bounds__(128) void ppo_critic_kernel(
    const float* __restrict__ states, const float* __restrict__ next_states,
    const float* __restrict__ cW1, const float* __restrict__ cb1,
    const float* __restrict__ cW2, const float* __restrict__ cb2,
    const float* __restrict__ cW3, const float* __restrict__ cb3,
    float* __restrict__ vals, float* __restrict__ nvals, int T) {
  __shared__ __align__(16) unsigned short fW1[16 * 512];
  __shared__ __align__(16) unsigned short fW2[8 * 512];
  __shared__ float sb1[64], sb2[64], sw3[64];
  __shared__ __align__(16) unsigned short sX[WAVES][16 * LDX];
  __shared__ __align__(16) unsigned short sH1[WAVES][16 * LDH];
  int tid = threadIdx.x;
  prep_frags(cW1, fW1, 4, 4, 64, tid, 128);
  prep_frags(cW2, fW2, 2, 4, 64, tid, 128);
  if (tid < 64) { sb1[tid] = cb1[tid]; sb2[tid] = cb2[tid]; sw3[tid] = cW3[tid]; }
  __syncthreads();
  float b3 = cb3[0];
  int w = tid >> 5;
  int lane = tid & 31;
  int numTiles = (T + 15) >> 4;
  for (int tile = blockIdx.x * WAVES + w; tile < numTiles;
       tile += gridDim.x * WAVES) {
    int t0 = tile * 16;
    int pt0 = (tile + gridDim.x * WAVES) * 16;
    if (pt0 < T) {
      __builtin_prefetch(states + (size_t)pt0 * SDIM + lane * 4, 0, 0);
      __builtin_prefetch(next_states + (size_t)pt0 * SDIM + lane * 4, 0, 0);
    }
    critic_tile(states, vals, t0, T, fW1, fW2, sb1, sb2, sw3, b3, sX[w], sH1[w]);
    critic_tile(next_states, nvals, t0, T, fW1, fW2, sb1, sb2, sw3, b3, sX[w], sH1[w]);
  }
}

// ---------------- actor: logp(a_t), per-row entropy via WMMA MLP ----------------
__global__ __launch_bounds__(128) void ppo_actor_kernel(
    const float* __restrict__ states, const int* __restrict__ actions,
    const float* __restrict__ aW1, const float* __restrict__ ab1,
    const float* __restrict__ aW2, const float* __restrict__ ab2,
    const float* __restrict__ aW3, const float* __restrict__ ab3,
    float* __restrict__ logp_out, float* __restrict__ entPart, int T) {
  __shared__ __align__(16) unsigned short fW1[16 * 512];
  __shared__ __align__(16) unsigned short fW2[8 * 512];
  __shared__ __align__(16) unsigned short fW3[2 * 512];
  __shared__ float sb1[64], sb2[64], sb3[16];
  __shared__ __align__(16) unsigned short sX[WAVES][16 * LDX];
  __shared__ __align__(16) unsigned short sH1[WAVES][16 * LDH];
  __shared__ float sr[128];
  int tid = threadIdx.x;
  prep_frags(aW1, fW1, 4, 4, 64, tid, 128);
  prep_frags(aW2, fW2, 2, 4, 64, tid, 128);
  prep_frags(aW3, fW3, 2, 1, 16, tid, 128);
  if (tid < 64) { sb1[tid] = ab1[tid]; sb2[tid] = ab2[tid]; }
  if (tid < 16) sb3[tid] = ab3[tid];
  __syncthreads();
  int w = tid >> 5;
  int lane = tid & 31;
  int n = lane & 15;
  int hi = lane >> 4;
  float entAcc = 0.f;
  int numTiles = (T + 15) >> 4;
  for (int tile = blockIdx.x * WAVES + w; tile < numTiles;
       tile += gridDim.x * WAVES) {
    int t0 = tile * 16;
    int pt0 = (tile + gridDim.x * WAVES) * 16;
    if (pt0 < T) __builtin_prefetch(states + (size_t)pt0 * SDIM + lane * 4, 0, 0);
    unsigned short* sXw = sX[w];
    unsigned short* sH1w = sH1[w];
    unsigned short* H2 = sXw;  // reuse X staging once consumed by L1
    stage_rows(states, t0, T, sXw);
    v16bf aX[4];
    #pragma unroll
    for (int kc = 0; kc < 4; ++kc) aX[kc] = load_a(sXw, LDX, kc * 32);
    // L1
    #pragma unroll
    for (int nt = 0; nt < 4; ++nt) {
      v8f acc = {};
      #pragma unroll
      for (int kc = 0; kc < 4; ++kc)
        acc = wmma_bf16(aX[kc], load_b(fW1 + (kc * 4 + nt) * 512), acc);
      float b = sb1[nt * 16 + n];
      #pragma unroll
      for (int r = 0; r < 8; ++r)
        sH1w[(r + hi * 8) * LDH + nt * 16 + n] = f2bf(fmaxf(acc[r] + b, 0.f));
    }
    v16bf aH[2];
    #pragma unroll
    for (int kc = 0; kc < 2; ++kc) aH[kc] = load_a(sH1w, LDH, kc * 32);
    // L2
    #pragma unroll
    for (int nt = 0; nt < 4; ++nt) {
      v8f acc = {};
      #pragma unroll
      for (int kc = 0; kc < 2; ++kc)
        acc = wmma_bf16(aH[kc], load_b(fW2 + (kc * 4 + nt) * 512), acc);
      float b = sb2[nt * 16 + n];
      #pragma unroll
      for (int r = 0; r < 8; ++r)
        H2[(r + hi * 8) * LDH + nt * 16 + n] = f2bf(fmaxf(acc[r] + b, 0.f));
    }
    v16bf aH2[2];
    #pragma unroll
    for (int kc = 0; kc < 2; ++kc) aH2[kc] = load_a(H2, LDH, kc * 32);
    // L3: logits (16x16)
    v8f acc = {};
    #pragma unroll
    for (int kc = 0; kc < 2; ++kc)
      acc = wmma_bf16(aH2[kc], load_b(fW3 + kc * 512), acc);
    // softmax / entropy / logp over the 16 actions held across a 16-lane group
    #pragma unroll
    for (int r = 0; r < 8; ++r) {
      int t = t0 + r + hi * 8;
      float l = acc[r] + sb3[n];
      float mx = l;
      #pragma unroll
      for (int m = 1; m <= 8; m <<= 1) mx = fmaxf(mx, __shfl_xor(mx, m, 32));
      float e = __expf(l - mx);
      float se = e;
      #pragma unroll
      for (int m = 1; m <= 8; m <<= 1) se += __shfl_xor(se, m, 32);
      float p = e / se;
      float ec = -p * __logf(p + 1e-8f);
      #pragma unroll
      for (int m = 1; m <= 8; m <<= 1) ec += __shfl_xor(ec, m, 32);
      int act = (t < T) ? actions[t] : 0;
      float lp = (n == act) ? __logf(p) : 0.f;
      #pragma unroll
      for (int m = 1; m <= 8; m <<= 1) lp += __shfl_xor(lp, m, 32);
      if (n == 0 && t < T) {
        logp_out[t] = lp;
        entAcc += ec;
      }
    }
  }
  // deterministic per-block entropy partial
  sr[tid] = entAcc;
  __syncthreads();
  for (int s = 64; s > 0; s >>= 1) {
    if (tid < s) sr[tid] += sr[tid + s];
    __syncthreads();
  }
  if (tid == 0) entPart[blockIdx.x] = sr[0];
}

// ---------------- GAE: segmented reverse affine scan ----------------
__global__ __launch_bounds__(256) void ppo_gae_seg(
    const float* __restrict__ rewards, const float* __restrict__ dones,
    const float* __restrict__ vals, const float* __restrict__ nvals,
    float* __restrict__ segC, float* __restrict__ segD, int T, int nseg) {
  int s = blockIdx.x * blockDim.x + threadIdx.x;
  if (s >= nseg) return;
  int start = s * SEG;
  int end = min(start + SEG, T);
  float A = 0.f, C = 1.f;
  for (int t = end - 1; t >= start; --t) {
    float nd = 1.f - dones[t];
    float c = GAMMA_F * LAM_F * nd;
    float d = rewards[t] + GAMMA_F * nvals[t] * nd - vals[t];
    A = c * A + d;
    C *= c;
  }
  segC[s] = C;
  segD[s] = A;
}

__global__ void ppo_gae_scan(const float* __restrict__ segC,
                             const float* __restrict__ segD,
                             float* __restrict__ carry, int nseg) {
  if (blockIdx.x == 0 && threadIdx.x == 0) {
    float x = 0.f;
    for (int s = nseg - 1; s >= 0; --s) {
      carry[s] = x;               // advantage entering segment s from the right
      x = segC[s] * x + segD[s];  // advantage at segment s left boundary
    }
  }
}

__global__ __launch_bounds__(256) void ppo_gae_apply(
    const float* __restrict__ rewards, const float* __restrict__ dones,
    const float* __restrict__ vals, const float* __restrict__ nvals,
    const float* __restrict__ carry, float* __restrict__ adv, int T, int nseg) {
  int s = blockIdx.x * blockDim.x + threadIdx.x;
  if (s >= nseg) return;
  int start = s * SEG;
  int end = min(start + SEG, T);
  float A = carry[s];
  for (int t = end - 1; t >= start; --t) {
    float nd = 1.f - dones[t];
    float c = GAMMA_F * LAM_F * nd;
    float d = rewards[t] + GAMMA_F * nvals[t] * nd - vals[t];
    A = c * A + d;
    adv[t] = A;
  }
}

// ---------------- reductions / finalize ----------------
__global__ __launch_bounds__(256) void ppo_adv_reduce(const float* __restrict__ adv,
                                                      int T, double* __restrict__ parts) {
  __shared__ double sd[256];
  double s = 0.0, sq = 0.0;
  for (int t = blockIdx.x * 256 + threadIdx.x; t < T; t += gridDim.x * 256) {
    double a = adv[t];
    s += a;
    sq += a * a;
  }
  s = block_reduce_d(s, sd);
  sq = block_reduce_d(sq, sd);
  if (threadIdx.x == 0) {
    parts[2 * blockIdx.x] = s;
    parts[2 * blockIdx.x + 1] = sq;
  }
}

__global__ __launch_bounds__(256) void ppo_stats(const double* __restrict__ parts,
                                                 int nb, int T, float* __restrict__ scal) {
  __shared__ double sd[256];
  double s = 0.0, sq = 0.0;
  for (int i = threadIdx.x; i < nb; i += 256) {
    s += parts[2 * i];
    sq += parts[2 * i + 1];
  }
  s = block_reduce_d(s, sd);
  sq = block_reduce_d(sq, sd);
  if (threadIdx.x == 0) {
    double mean = s / (double)T;
    double var = (sq - s * s / (double)T) / (double)(T - 1);
    if (var < 0.0) var = 0.0;
    scal[0] = (float)mean;
    scal[1] = (float)(1.0 / (sqrt(var) + 1e-8));
    scal[2] = (float)(sq / (double)T);  // critic loss: values-returns == -adv
  }
}

__global__ __launch_bounds__(256) void ppo_actor_reduce(
    const float* __restrict__ adv, const float* __restrict__ logp,
    const float* __restrict__ oldlp, const float* __restrict__ scal, int T,
    double* __restrict__ parts) {
  __shared__ double sd[256];
  float mean = scal[0], inv = scal[1];
  double s = 0.0;
  for (int t = blockIdx.x * 256 + threadIdx.x; t < T; t += gridDim.x * 256) {
    float a = (adv[t] - mean) * inv;
    float ratio = __expf(logp[t] - oldlp[t]);
    float rc = fminf(fmaxf(ratio, 1.0f - CLIP_F), 1.0f + CLIP_F);
    s += (double)fminf(ratio * a, rc * a);
  }
  s = block_reduce_d(s, sd);
  if (threadIdx.x == 0) parts[blockIdx.x] = s;
}

__global__ __launch_bounds__(256) void ppo_finalize(
    const double* __restrict__ partMin, int nbMin,
    const float* __restrict__ entPart, int nbe,
    const float* __restrict__ scal, int T, float* __restrict__ out) {
  __shared__ double sd[256];
  double s = 0.0;
  for (int i = threadIdx.x; i < nbMin; i += 256) s += partMin[i];
  double smin = block_reduce_d(s, sd);
  double e = 0.0;
  for (int i = threadIdx.x; i < nbe; i += 256) e += (double)entPart[i];
  e = block_reduce_d(e, sd);
  if (threadIdx.x == 0) {
    float actor = -(float)(smin / (double)T);
    float critic = scal[2];
    float ent = (float)(e / (double)T);
    float total = actor + 0.5f * critic - 0.01f * ent;
    out[0] = total;
    out[1] = actor;
    out[2] = critic;
    out[3] = ent;
  }
}

// ---------------- host launcher ----------------
extern "C" void kernel_launch(void* const* d_in, const int* in_sizes, int n_in,
                              void* d_out, int out_size, void* d_ws, size_t ws_size,
                              hipStream_t stream) {
  const float* states      = (const float*)d_in[0];
  const float* next_states = (const float*)d_in[1];
  const float* rewards     = (const float*)d_in[2];
  const float* dones       = (const float*)d_in[3];
  const int*   actions     = (const int*)d_in[4];
  const float* old_logp    = (const float*)d_in[5];
  const float* aW1 = (const float*)d_in[6];
  const float* ab1 = (const float*)d_in[7];
  const float* aW2 = (const float*)d_in[8];
  const float* ab2 = (const float*)d_in[9];
  const float* aW3 = (const float*)d_in[10];
  const float* ab3 = (const float*)d_in[11];
  const float* cW1 = (const float*)d_in[12];
  const float* cb1 = (const float*)d_in[13];
  const float* cW2 = (const float*)d_in[14];
  const float* cb2 = (const float*)d_in[15];
  const float* cW3 = (const float*)d_in[16];
  const float* cb3 = (const float*)d_in[17];

  int T = in_sizes[2];  // rewards length
  int NSEG = (T + SEG - 1) / SEG;

  char* base = (char*)d_ws;
  size_t off = 0;
  auto alloc = [&](size_t bytes) -> void* {
    off = (off + 255) & ~(size_t)255;
    void* p = base + off;
    off += bytes;
    return p;
  };
  float*  vals  = (float*)alloc(sizeof(float) * (size_t)T);
  float*  nvals = (float*)alloc(sizeof(float) * (size_t)T);
  float*  logp  = (float*)alloc(sizeof(float) * (size_t)T);
  float*  adv   = (float*)alloc(sizeof(float) * (size_t)T);
  float*  segC  = (float*)alloc(sizeof(float) * (size_t)NSEG);
  float*  segD  = (float*)alloc(sizeof(float) * (size_t)NSEG);
  float*  carry = (float*)alloc(sizeof(float) * (size_t)NSEG);
  float*  entP  = (float*)alloc(sizeof(float) * GA);
  double* part1 = (double*)alloc(sizeof(double) * 2 * RB);
  double* part2 = (double*)alloc(sizeof(double) * RB2);
  float*  scal  = (float*)alloc(sizeof(float) * 8);
  (void)ws_size; (void)n_in; (void)out_size;

  ppo_actor_kernel<<<GA, 128, 0, stream>>>(states, actions, aW1, ab1, aW2, ab2,
                                           aW3, ab3, logp, entP, T);
  ppo_critic_kernel<<<GC, 128, 0, stream>>>(states, next_states, cW1, cb1, cW2,
                                            cb2, cW3, cb3, vals, nvals, T);
  ppo_gae_seg<<<(NSEG + 255) / 256, 256, 0, stream>>>(rewards, dones, vals,
                                                      nvals, segC, segD, T, NSEG);
  ppo_gae_scan<<<1, 32, 0, stream>>>(segC, segD, carry, NSEG);
  ppo_gae_apply<<<(NSEG + 255) / 256, 256, 0, stream>>>(rewards, dones, vals,
                                                        nvals, carry, adv, T, NSEG);
  ppo_adv_reduce<<<RB, 256, 0, stream>>>(adv, T, part1);
  ppo_stats<<<1, 256, 0, stream>>>(part1, RB, T, scal);
  ppo_actor_reduce<<<RB2, 256, 0, stream>>>(adv, logp, old_logp, scal, T, part2);
  ppo_finalize<<<1, 256, 0, stream>>>(part2, RB2, entP, GA, scal, T, (float*)d_out);
}
